// FactoredAttention_57011395887108
// MI455X (gfx1250) — matrix operand order repaired and checked by reading, not compiled, not run on hardware
//
#include <hip/hip_runtime.h>

// BS=2, L=8192, D=1024, H=16, BLOCKS=16 -> BC=512, HD=64, M=BS*L=16384
// SCALE2 = 1/sqrt(64) = 0.125

typedef __attribute__((ext_vector_type(16))) __bf16 v16bf;
typedef __attribute__((ext_vector_type(8)))  float v8f;

typedef unsigned int u32x4 __attribute__((ext_vector_type(4)));
typedef int          i32x8 __attribute__((ext_vector_type(8)));
typedef int          i32x4 __attribute__((ext_vector_type(4)));

#if defined(__has_builtin)
#if __has_builtin(__builtin_amdgcn_tensor_load_to_lds) && \
    __has_builtin(__builtin_amdgcn_s_wait_tensorcnt)
#define HAS_TDM 1
#endif
#if __has_builtin(__builtin_amdgcn_global_load_async_to_lds_b128)
#define HAS_ASYNC 1
#endif
#endif

__device__ __forceinline__ void wait_async0() {
#if defined(__has_builtin) && __has_builtin(__builtin_amdgcn_s_wait_asynccnt)
  __builtin_amdgcn_s_wait_asynccnt(0);
#else
  asm volatile("s_wait_asynccnt 0x0" ::: "memory");
#endif
}

union FragBF {
  v16bf v;
  unsigned int u[8];
  unsigned short s16[16];
};

__device__ __forceinline__ unsigned short f2bf(float f) {
  unsigned int u = __float_as_uint(f);
  u += 0x7FFFu + ((u >> 16) & 1u);   // round-to-nearest-even
  return (unsigned short)(u >> 16);
}

// ---------------------------------------------------------------------------
// GEMM: C[M,N] = A[M,K] (f32 or bf16) * B[K,N] (f32, converted bf16) + bias
// 128x128 block tile, 8 waves (4 along M x 2 along N), each wave 32x64,
// K-step 32 -> one v_wmma_f32_16x16x32_bf16 per 16x16 tile per step.
// When A is bf16, the A tile is staged with global_load_async_to_lds_b128.
// ---------------------------------------------------------------------------
template <bool A_BF16, bool OUT_BF16>
__global__ __launch_bounds__(256) void gemm_bias_wmma(
    const void* __restrict__ Ap, const float* __restrict__ B,
    const float* __restrict__ bias, void* __restrict__ Cp,
    int M, int N, int K)
{
  __shared__ unsigned short As[128 * 32];  // [m][k] bf16 bits (8KB)
  __shared__ unsigned short Bs[128 * 32];  // [n][k] bf16 bits (8KB)

  const int tid  = threadIdx.x;
  const int wave = tid >> 5, lane = tid & 31;
  const int hx = lane >> 4, lm = lane & 15;
  const int waveM = wave & 3, waveN = wave >> 2;
  const int m0 = blockIdx.y * 128, n0 = blockIdx.x * 128;

  v8f acc[2][4];
#pragma unroll
  for (int i = 0; i < 2; ++i)
#pragma unroll
    for (int j = 0; j < 4; ++j) acc[i][j] = {};

  for (int k0 = 0; k0 < K; k0 += 32) {
    // ---- stage A tile (128x32) as bf16 ----
    if constexpr (A_BF16) {
      const unsigned short* A = (const unsigned short*)Ap;
#if defined(HAS_ASYNC)
      // async: each lane copies one 16B chunk (8 bf16), 512 chunks total
#pragma unroll
      for (int it = 0; it < 2; ++it) {
        int e = tid + it * 256;
        int m = e >> 2, c8 = (e & 3) * 8;
        __builtin_amdgcn_global_load_async_to_lds_b128(
            (__attribute__((address_space(1))) i32x4*)(A + (size_t)(m0 + m) * K + k0 + c8),
            (__attribute__((address_space(3))) i32x4*)&As[m * 32 + c8], 0, 0);
      }
#else
#pragma unroll
      for (int it = 0; it < 4; ++it) {
        int e = tid + it * 256;
        int m = e >> 3, kg = (e & 7) * 4;
        uint2 val = *(const uint2*)(A + (size_t)(m0 + m) * K + k0 + kg);
        *(uint2*)&As[m * 32 + kg] = val;
      }
#endif
    } else {
#pragma unroll
      for (int it = 0; it < 4; ++it) {
        int e = tid + it * 256;             // 1024 groups of 4 elements
        int m = e >> 3, kg = (e & 7) * 4;
        const float* A = (const float*)Ap;
        const float* src = A + (size_t)(m0 + m) * K + k0 + kg;
        if (k0 + 32 < K) __builtin_prefetch(src + 32, 0, 1);
        float4 f = *(const float4*)src;
        uint2 val;
        val.x = (unsigned)f2bf(f.x) | ((unsigned)f2bf(f.y) << 16);
        val.y = (unsigned)f2bf(f.z) | ((unsigned)f2bf(f.w) << 16);
        *(uint2*)&As[m * 32 + kg] = val;
      }
    }
    // ---- stage B tile (32x128) transposed -> Bs[n][k] (f32 -> bf16) ----
#pragma unroll
    for (int it = 0; it < 4; ++it) {
      int e = tid + it * 256;
      int k = e >> 5, ng = (e & 31) * 4;
      float4 f = *(const float4*)(B + (size_t)(k0 + k) * N + n0 + ng);
      Bs[(ng + 0) * 32 + k] = f2bf(f.x);
      Bs[(ng + 1) * 32 + k] = f2bf(f.y);
      Bs[(ng + 2) * 32 + k] = f2bf(f.z);
      Bs[(ng + 3) * 32 + k] = f2bf(f.w);
    }
#if defined(HAS_ASYNC)
    if constexpr (A_BF16) wait_async0();
#endif
    __syncthreads();

    // ---- per-lane fragment gather (contiguous k = 16*hx + s, same for A and B) ----
    FragBF a[2], b[4];
#pragma unroll
    for (int mt = 0; mt < 2; ++mt) {
      int r = waveM * 32 + mt * 16 + lm;
#pragma unroll
      for (int t = 0; t < 8; ++t) {
        int K0 = hx * 16 + 2 * t;
        a[mt].u[t] = *(const unsigned int*)&As[r * 32 + K0];
      }
    }
#pragma unroll
    for (int nt = 0; nt < 4; ++nt) {
      int c = waveN * 64 + nt * 16 + lm;
#pragma unroll
      for (int t = 0; t < 8; ++t) {
        int K0 = hx * 16 + 2 * t;
        b[nt].u[t] = *(const unsigned int*)&Bs[c * 32 + K0];
      }
    }
#pragma unroll
    for (int mt = 0; mt < 2; ++mt)
#pragma unroll
      for (int nt = 0; nt < 4; ++nt)
        acc[mt][nt] = __builtin_amdgcn_wmma_f32_16x16x32_bf16(
            false, a[mt].v, false, b[nt].v, (short)0, acc[mt][nt], false, false);
    __syncthreads();
  }

  // ---- epilogue: bias + store (C/D layout: M = v + 8*(lane/16), N = lane%16)
#pragma unroll
  for (int mt = 0; mt < 2; ++mt)
#pragma unroll
    for (int nt = 0; nt < 4; ++nt) {
      int col = n0 + waveN * 64 + nt * 16 + lm;
      float bb = bias[col];
#pragma unroll
      for (int v = 0; v < 8; ++v) {
        int row = m0 + waveM * 32 + mt * 16 + v + 8 * hx;
        float x = acc[mt][nt][v] + bb;
        if constexpr (OUT_BF16)
          ((unsigned short*)Cp)[(size_t)row * N + col] = f2bf(x);
        else
          ((float*)Cp)[(size_t)row * N + col] = x;
      }
    }
}

// ---------------------------------------------------------------------------
// Block-local causal attention, flash-style online softmax.
// grid = 512 blocks: one (batch-block nb in 0..31, head h in 0..15) each.
// K[512,64] staged by the Tensor Data Mover (2D D#: 64-wide rows, stride
// 3072, 512 rows); V^T[64,512] staged manually; 8 waves x 64 query rows.
// ---------------------------------------------------------------------------
__global__ __launch_bounds__(256) void attn_wmma(
    const unsigned short* __restrict__ qkv, unsigned short* __restrict__ aout)
{
  extern __shared__ unsigned short smem[];
  unsigned short* Ks  = smem;                 // [512][64]  64KB
  unsigned short* Vst = smem + 512 * 64;      // [64][512]  64KB (V transposed)
  unsigned short* Ps  = Vst + 64 * 512;       // [8][16][32] 8KB P scratch

  const int tid  = threadIdx.x;
  const int wave = tid >> 5, lane = tid & 31;
  const int hx = lane >> 4, lm = lane & 15;
  const int h  = blockIdx.x & 15;
  const int nb = blockIdx.x >> 4;
  const long tokbase = (long)nb * 512;
  const unsigned int* qkv_u = (const unsigned int*)qkv;
  const float SCALE2 = 0.125f;                // 1/sqrt(64)

#if defined(HAS_TDM)
  if (wave == 0) {
    // --- Tensor DMA descriptor: K tile 512x64 bf16, row stride 3072 elems ---
    unsigned long long ga = (unsigned long long)(const void*)qkv +
        ((unsigned long long)tokbase * 3072ull + 1024ull + (unsigned long long)h * 64ull) * 2ull;
    unsigned lds_off = (unsigned)(__SIZE_TYPE__)
        (__attribute__((address_space(3))) unsigned short*)Ks;
    u32x4 g0;
    g0.x = 1u;                                        // count=1, user mode
    g0.y = lds_off;                                   // lds_addr (bytes)
    g0.z = (unsigned)(ga & 0xFFFFFFFFu);              // global_addr[31:0]
    g0.w = (unsigned)((ga >> 32) & 0x01FFFFFFu)       // global_addr[56:32]
           | 0x80000000u;                             // type=2 ("image")
    i32x8 g1;
    g1[0] = 0x10000;                                  // data_size=1 (2 bytes)
    g1[1] = (int)(64u << 16);                         // tensor_dim0 = 64
    g1[2] = (int)(512u << 16);                        // tensor_dim1 = 512
    g1[3] = (int)(64u << 16);                         // tile_dim0 = 64
    g1[4] = 512;                                      // tile_dim1 = 512
    g1[5] = 3072;                                     // tensor_dim0_stride
    g1[6] = 0;
    g1[7] = 0;
    i32x4 g2 = {0, 0, 0, 0};
    i32x4 g3 = {0, 0, 0, 0};
    i32x8 g4 = {0, 0, 0, 0, 0, 0, 0, 0};
    __builtin_amdgcn_tensor_load_to_lds(g0, g1, g2, g3, g4, 0);
    __builtin_amdgcn_s_wait_tensorcnt(0);
  }
#else
  // stage K rows (uint = 2 bf16)
  for (int i = tid; i < 512 * 32; i += 256) {
    int t = i >> 5, d2 = i & 31;
    size_t src = ((size_t)(tokbase + t) * 3072 + 1024 + h * 64) >> 1;
    *(unsigned int*)&Ks[t * 64 + d2 * 2] = qkv_u[src + d2];
  }
#endif
  // stage V transposed: Vst[d][t]
  for (int i = tid; i < 512 * 64; i += 256) {
    int t = i >> 6, d = i & 63;
    Vst[d * 512 + t] = qkv[(size_t)(tokbase + t) * 3072 + 2048 + h * 64 + d];
  }
  __syncthreads();

  unsigned short* Pw = Ps + wave * (16 * 32);

  for (int qt = 0; qt < 4; ++qt) {
    const int qrow0 = wave * 64 + qt * 16;

    // Q fragments (16 rows x 64 dims = two 16x32 A-frags) straight from global
    FragBF aq[2];
    {
      size_t qb = ((size_t)(tokbase + qrow0 + lm) * 3072 + h * 64) >> 1;
#pragma unroll
      for (int ks = 0; ks < 2; ++ks)
#pragma unroll
        for (int t = 0; t < 8; ++t) {
          int K0 = ks * 32 + hx * 16 + 2 * t;
          aq[ks].u[t] = qkv_u[qb + (K0 >> 1)];
        }
    }

    float mrow[8], lrow[8], alpha8[8];
    v8f o[4];
#pragma unroll
    for (int v = 0; v < 8; ++v) { mrow[v] = -1e30f; lrow[v] = 0.0f; }
#pragma unroll
    for (int nt = 0; nt < 4; ++nt) o[nt] = {};

    const int npairs = (qrow0 / 32) + 1;   // 32-key steps covering causal range
    for (int p = 0; p < npairs; ++p) {
      const int j0 = p * 32;

      // ---- scores: S[16,32] = Q[16,64] * K^T, two 16x16 tiles ----
      v8f s[2];
#pragma unroll
      for (int sub = 0; sub < 2; ++sub) {
        v8f st = {};
        const int key = j0 + sub * 16 + lm;     // B-frag column = key index
#pragma unroll
        for (int ks = 0; ks < 2; ++ks) {
          FragBF bk;
#pragma unroll
          for (int t = 0; t < 8; ++t)
            bk.u[t] = *(const unsigned int*)&Ks[key * 64 + ks * 32 + hx * 16 + 2 * t];
          st = __builtin_amdgcn_wmma_f32_16x16x32_bf16(
              false, aq[ks].v, false, bk.v, (short)0, st, false, false);
        }
        s[sub] = st;
      }

      // ---- scale, causal mask, online softmax update ----
#pragma unroll
      for (int v = 0; v < 8; ++v) {
        const int qr = qrow0 + v + 8 * hx;
        float s0 = s[0][v] * SCALE2;
        float s1 = s[1][v] * SCALE2;
        if (j0 + lm > qr)      s0 = -1e30f;
        if (j0 + 16 + lm > qr) s1 = -1e30f;
        float mx = fmaxf(s0, s1);
#pragma unroll
        for (int off = 1; off < 16; off <<= 1)
          mx = fmaxf(mx, __shfl_xor(mx, off, 32));   // row reduce (16 lanes/half)
        const float mnew = fmaxf(mrow[v], mx);
        const float al = __expf(mrow[v] - mnew);
        const float p0 = __expf(s0 - mnew);
        const float p1 = __expf(s1 - mnew);
        float rs = p0 + p1;
#pragma unroll
        for (int off = 1; off < 16; off <<= 1)
          rs += __shfl_xor(rs, off, 32);
        lrow[v] = lrow[v] * al + rs;
        mrow[v] = mnew;
        alpha8[v] = al;
        s[0][v] = p0;
        s[1][v] = p1;
      }
#pragma unroll
      for (int nt = 0; nt < 4; ++nt)
#pragma unroll
        for (int v = 0; v < 8; ++v)
          o[nt][v] = o[nt][v] * alpha8[v];

      // ---- repack P (accumulator layout) -> A-fragment layout via LDS ----
#pragma unroll
      for (int v = 0; v < 8; ++v) {
        const int m = v + 8 * hx;
        Pw[m * 32 + lm]      = f2bf(s[0][v]);
        Pw[m * 32 + 16 + lm] = f2bf(s[1][v]);
      }
      FragBF ap;  // same-wave LDS ops are in-order; compiler inserts s_wait_dscnt
#pragma unroll
      for (int t = 0; t < 8; ++t) {
        int K0 = hx * 16 + 2 * t;
        ap.u[t] = *(const unsigned int*)&Pw[lm * 32 + K0];
      }

      // ---- O += P[16,32] * V[32,64] : 4 WMMAs ----
#pragma unroll
      for (int nt = 0; nt < 4; ++nt) {
        FragBF bv;
        const int d = nt * 16 + lm;
#pragma unroll
        for (int t = 0; t < 8; ++t)
          bv.u[t] = *(const unsigned int*)&Vst[d * 512 + j0 + hx * 16 + 2 * t];
        o[nt] = __builtin_amdgcn_wmma_f32_16x16x32_bf16(
            false, ap.v, false, bv.v, (short)0, o[nt], false, false);
      }
    }

    // ---- normalize and write attention output tile (bf16) ----
#pragma unroll
    for (int v = 0; v < 8; ++v) alpha8[v] = 1.0f / lrow[v];
#pragma unroll
    for (int nt = 0; nt < 4; ++nt) {
      const int col = h * 64 + nt * 16 + lm;
#pragma unroll
      for (int v = 0; v < 8; ++v) {
        const long row = tokbase + qrow0 + v + 8 * hx;
        aout[(size_t)row * 1024 + col] = f2bf(o[nt][v] * alpha8[v]);
      }
    }
  }
}

// ---------------------------------------------------------------------------
extern "C" void kernel_launch(void* const* d_in, const int* in_sizes, int n_in,
                              void* d_out, int out_size, void* d_ws, size_t ws_size,
                              hipStream_t stream) {
  (void)in_sizes; (void)n_in; (void)out_size; (void)ws_size;
  const float* x      = (const float*)d_in[0];
  const float* w_attn = (const float*)d_in[1];
  const float* b_attn = (const float*)d_in[2];
  const float* w_proj = (const float*)d_in[3];
  const float* b_proj = (const float*)d_in[4];
  float* out = (float*)d_out;

  // workspace: qkv bf16 [16384,3072] (96MB) + attn out bf16 [16384,1024] (32MB)
  unsigned short* qkv  = (unsigned short*)d_ws;
  unsigned short* abuf = qkv + (size_t)16384 * 3072;

  dim3 blk(256);

  // 1) qkv = x @ w_attn + b_attn   (M=16384, N=3072, K=1024), bf16 out
  gemm_bias_wmma<false, true><<<dim3(3072 / 128, 16384 / 128), blk, 0, stream>>>(
      x, w_attn, b_attn, qkv, 16384, 3072, 1024);

  // 2) block-local causal attention (512 head-blocks), bf16 out
  size_t smem = (size_t)(512 * 64 + 64 * 512 + 8 * 16 * 32) * sizeof(unsigned short);
  attn_wmma<<<dim3(512), blk, smem, stream>>>(qkv, abuf);

  // 3) out = a @ w_proj + b_proj   (M=16384, N=1024, K=1024), f32 out
  gemm_bias_wmma<true, false><<<dim3(1024 / 128, 16384 / 128), blk, 0, stream>>>(
      abuf, w_proj, b_proj, out, 16384, 1024, 1024);
}
